// Freq_VQVAE_66984309948644
// MI455X (gfx1250) — matrix operand order repaired and checked by reading, not compiled, not run on hardware
//
#include <hip/hip_runtime.h>
#include <hip/hip_bf16.h>

// ---------------------------------------------------------------------------
// Freq-VQVAE forward on MI455X (gfx1250, wave32, WMMA).
// All GEMMs run on v_wmma_f32_16x16x32_bf16 (f32 accumulate). Weights are
// converted once to transposed [N,K] bf16 so B fragments are contiguous
// 32-byte lane loads; activation producers (LN / gelu / attention-out) emit
// bf16 so A fragments are contiguous 16-byte lane loads. Everything fits in
// the 192MB L2, so the design minimizes passes over activations.
// ---------------------------------------------------------------------------

#define BQ    16
#define NFQ   320
#define SQ    1024
#define DQ    64
#define HIDQ  256
#define LENCQ 12
#define LDECQ 3
#define CBQ   4096
#define NHQ   8
#define MQ    (BQ * SQ)   // 16384 tokens

typedef __attribute__((ext_vector_type(16))) __bf16 v16bf;
typedef __attribute__((ext_vector_type(8)))  __bf16 v8bf;
typedef __attribute__((ext_vector_type(8)))  float  v8f;

__device__ __forceinline__ __bf16 f2bf(float f) {
  union { float f; unsigned u; } in; in.f = f;
  unsigned u = in.u;
  u += 0x7FFFu + ((u >> 16) & 1u);              // round-to-nearest-even
  union { unsigned short s; __bf16 b; } out; out.s = (unsigned short)(u >> 16);
  return out.b;
}

__device__ __forceinline__ float gelu_f(float v) {
  float x3 = v * v * v;
  return 0.5f * v * (1.f + tanhf(0.7978845608028654f * (v + 0.044715f * x3)));
}

// ---------------------------------------------------------------------------
// Weight convert + transpose: w [L,K,N] f32 -> wt [L,N,K] bf16
// ---------------------------------------------------------------------------
__global__ void k_convw(const float* __restrict__ w, __bf16* __restrict__ wt,
                        int K, int N, long total) {
  long g = (long)blockIdx.x * blockDim.x + threadIdx.x;
  if (g >= total) return;
  int  n = (int)(g % N);
  long t = g / N;
  int  k = (int)(t % K);
  long l = t / K;
  wt[(l * N + n) * (long)K + k] = f2bf(w[g]);
}

// f32 -> bf16 (same layout), exact-sized launch
__global__ void k_convact(const float* __restrict__ s, __bf16* __restrict__ d) {
  long g = (long)blockIdx.x * blockDim.x + threadIdx.x;
  d[g] = f2bf(s[g]);
}

// ---------------------------------------------------------------------------
// x [B,NF,S] f32 -> xt [B,S,NF] bf16  (LDS 32x32 tiled transpose)
// ---------------------------------------------------------------------------
__global__ void k_transpose_x(const float* __restrict__ x, __bf16* __restrict__ xt) {
  __shared__ float tile[32][33];
  int b  = blockIdx.z;
  int f0 = blockIdx.x * 32;
  int s0 = blockIdx.y * 32;
  int tx = threadIdx.x, ty = threadIdx.y;           // 32 x 8
  #pragma unroll
  for (int i = 0; i < 32; i += 8)
    tile[ty + i][tx] = x[(long)b * NFQ * SQ + (long)(f0 + ty + i) * SQ + (s0 + tx)];
  __syncthreads();
  #pragma unroll
  for (int i = 0; i < 32; i += 8)
    xt[(long)b * SQ * NFQ + (long)(s0 + ty + i) * NFQ + (f0 + tx)] = f2bf(tile[tx][ty + i]);
}

// ---------------------------------------------------------------------------
// WMMA GEMM: C[M,N] = A[M,K](bf16) @ Wt[N,K](bf16)^T  (+ epilogue)
// One 16x16 tile per wave, 4 waves/block, grid = (N/16, M/64).
// ---------------------------------------------------------------------------
enum { EPI_NONE = 0, EPI_BIAS, EPI_RESID, EPI_BIAS_GELU_BF16, EPI_BIAS_RESID, EPI_BIAS_TRANS };

template <int EPI, int K, int N>
__global__ __launch_bounds__(128)
void k_gemm(const __bf16* __restrict__ A, const __bf16* __restrict__ Wt,
            const float* __restrict__ bias, const float* resid, void* Cv) {
  const int lane = threadIdx.x & 31;
  const int wave = threadIdx.x >> 5;
  const int l15  = lane & 15;
  const int half = lane >> 4;                 // 0 or 1
  const int tn   = blockIdx.x * 16;
  const int tm   = (blockIdx.y * 4 + wave) * 16;

  // A fragment: lane owns row m=tm+l15; K-base = half*16, second block at +8.
  const __bf16* ap = A + (long)(tm + l15) * K + half * 16;
  // B fragment: lane owns col n=tn+l15; K-base = half*16 (contiguous in [N,K]).
  const __bf16* bp = Wt + (long)(tn + l15) * K + half * 16;

  v8f acc = {0.f, 0.f, 0.f, 0.f, 0.f, 0.f, 0.f, 0.f};
  #pragma unroll
  for (int k0 = 0; k0 < K; k0 += 32) {
    v8bf alo = *(const v8bf*)(ap + k0);
    v8bf ahi = *(const v8bf*)(ap + k0 + 8);
    v16bf af;
    #pragma unroll
    for (int j = 0; j < 8; j++) { af[j] = alo[j]; af[8 + j] = ahi[j]; }
    v16bf bfv = *(const v16bf*)(bp + k0);
    acc = __builtin_amdgcn_wmma_f32_16x16x32_bf16(false, af, false, bfv,
                                                  (short)0, acc, false, false);
  }

  const int n = tn + l15;
  float bv = 0.f;
  if constexpr (EPI == EPI_BIAS || EPI == EPI_BIAS_GELU_BF16 ||
                EPI == EPI_BIAS_RESID || EPI == EPI_BIAS_TRANS)
    bv = bias[n];

  #pragma unroll
  for (int r = 0; r < 8; r++) {
    int   m = tm + half * 8 + r;
    float v = acc[r] + bv;
    if constexpr (EPI == EPI_BIAS_GELU_BF16) {
      ((__bf16*)Cv)[(long)m * N + n] = f2bf(gelu_f(v));
    } else if constexpr (EPI == EPI_BIAS_TRANS) {
      int bb = m >> 10, ss = m & 1023;                 // m = b*S + s
      ((float*)Cv)[((long)bb * NFQ + n) * SQ + ss] = v;
    } else {
      long off = (long)m * N + n;
      if constexpr (EPI == EPI_RESID || EPI == EPI_BIAS_RESID) v += resid[off];
      ((float*)Cv)[off] = v;
    }
  }
}

// ---------------------------------------------------------------------------
// Add sinusoidal positional encoding in place (X: [M,64] f32)
// ---------------------------------------------------------------------------
__global__ void k_addpe(float* __restrict__ X) {
  int g = blockIdx.x * blockDim.x + threadIdx.x;    // exactly M*64 threads
  int m = g >> 6, d = g & 63;
  int spos = m & (SQ - 1);
  float freq = __expf(-0.1439115683121279f * (float)(d & ~1));
  float ang  = (float)spos * freq;
  X[g] += (d & 1) ? __cosf(ang) : __sinf(ang);
}

// ---------------------------------------------------------------------------
// LayerNorm (64 features, wave per row) -> bf16 output for GEMM A side
// ---------------------------------------------------------------------------
__global__ void k_layernorm(const float* __restrict__ X, const float* __restrict__ sc,
                            const float* __restrict__ bi, __bf16* __restrict__ Y) {
  int gw   = (blockIdx.x * blockDim.x + threadIdx.x) >> 5;   // row
  int lane = threadIdx.x & 31;
  const float* row = X + (long)gw * 64;
  float v0 = row[lane], v1 = row[lane + 32];
  float s = v0 + v1;
  #pragma unroll
  for (int o = 16; o > 0; o >>= 1) s += __shfl_xor(s, o, 32);
  float mean = s * (1.f / 64.f);
  float d0 = v0 - mean, d1 = v1 - mean;
  float vv = d0 * d0 + d1 * d1;
  #pragma unroll
  for (int o = 16; o > 0; o >>= 1) vv += __shfl_xor(vv, o, 32);
  float rinv = rsqrtf(vv * (1.f / 64.f) + 1e-5f);
  Y[(long)gw * 64 + lane]      = f2bf(d0 * rinv * sc[lane] + bi[lane]);
  Y[(long)gw * 64 + lane + 32] = f2bf(d1 * rinv * sc[lane + 32] + bi[lane + 32]);
}

// ---------------------------------------------------------------------------
// k softmax over sequence: wave per (b,d); QKV [M,192], k at col 64+d
// ---------------------------------------------------------------------------
__global__ void k_ksoftmax(const float* __restrict__ qkv, float* __restrict__ kb) {
  int gw   = (blockIdx.x * blockDim.x + threadIdx.x) >> 5;   // 0..B*64-1
  int lane = threadIdx.x & 31;
  int b = gw >> 6, d = gw & 63;
  const float* base = qkv + (long)b * SQ * 192 + 64 + d;
  float vals[32];
  float vmax = -3.4e38f;
  #pragma unroll
  for (int i = 0; i < 32; i++) {
    vals[i] = base[(long)(i * 32 + lane) * 192];
    vmax = fmaxf(vmax, vals[i]);
  }
  #pragma unroll
  for (int o = 16; o > 0; o >>= 1) vmax = fmaxf(vmax, __shfl_xor(vmax, o, 32));
  float sum = 0.f;
  #pragma unroll
  for (int i = 0; i < 32; i++) { vals[i] = __expf(vals[i] - vmax); sum += vals[i]; }
  #pragma unroll
  for (int o = 16; o > 0; o >>= 1) sum += __shfl_xor(sum, o, 32);
  float inv = 1.f / sum;
  float* out = kb + (long)b * SQ * 64 + d;
  #pragma unroll
  for (int i = 0; i < 32; i++) out[(long)(i * 32 + lane) * 64] = vals[i] * inv;
}

// ---------------------------------------------------------------------------
// ctx[b,h,d,e] = sum_s k[b,s,h,d] * v[b,s,h,e]   (thread per (b,h,d,e))
// ---------------------------------------------------------------------------
__global__ void k_ctx(const float* __restrict__ kb, const float* __restrict__ qkv,
                      float* __restrict__ ctx) {
  int t  = threadIdx.x;
  int bh = blockIdx.x * 4 + (t >> 6);               // 4 (b,h) pairs / block
  int b = bh >> 3, h = bh & 7;
  int d = (t >> 3) & 7, e = t & 7;
  const float* kp = kb  + (long)b * SQ * 64  + h * 8 + d;
  const float* vp = qkv + (long)b * SQ * 192 + 128 + h * 8 + e;
  float acc = 0.f;
  for (int s = 0; s < SQ; s++) acc += kp[(long)s * 64] * vp[(long)s * 192];
  ctx[(long)bh * 64 + d * 8 + e] = acc;
}

// ---------------------------------------------------------------------------
// per-token q softmax (over 8 dims/head) then o = q_sm @ ctx -> bf16 [M,64]
// ---------------------------------------------------------------------------
__global__ void k_qctx(const float* __restrict__ qkv, const float* __restrict__ ctx,
                       __bf16* __restrict__ O) {
  int g = blockIdx.x * blockDim.x + threadIdx.x;    // exactly M*8 threads
  int m = g >> 3, h = g & 7;
  int b = m >> 10;
  const float* q = qkv + (long)m * 192 + h * 8;
  float qv[8]; float mx = -3.4e38f;
  #pragma unroll
  for (int d = 0; d < 8; d++) { qv[d] = q[d]; mx = fmaxf(mx, qv[d]); }
  float sum = 0.f;
  #pragma unroll
  for (int d = 0; d < 8; d++) { qv[d] = __expf(qv[d] - mx); sum += qv[d]; }
  float inv = 1.f / sum;
  const float* c = ctx + ((long)b * 8 + h) * 64;
  float ov[8];
  #pragma unroll
  for (int e = 0; e < 8; e++) ov[e] = 0.f;
  #pragma unroll
  for (int d = 0; d < 8; d++) {
    float qq = qv[d] * inv;
    #pragma unroll
    for (int e = 0; e < 8; e++) ov[e] += qq * c[d * 8 + e];
  }
  #pragma unroll
  for (int e = 0; e < 8; e++) O[(long)m * 64 + h * 8 + e] = f2bf(ov[e]);
}

// ---------------------------------------------------------------------------
// L2 normalize rows of X in place + copy to quant_in region
// ---------------------------------------------------------------------------
__global__ void k_l2norm(float* __restrict__ X, float* __restrict__ qin) {
  int gw   = (blockIdx.x * blockDim.x + threadIdx.x) >> 5;
  int lane = threadIdx.x & 31;
  float v0 = X[(long)gw * 64 + lane], v1 = X[(long)gw * 64 + lane + 32];
  float ss = v0 * v0 + v1 * v1;
  #pragma unroll
  for (int o = 16; o > 0; o >>= 1) ss += __shfl_xor(ss, o, 32);
  float inv = 1.f / fmaxf(sqrtf(ss), 1e-12f);
  float a = v0 * inv, b = v1 * inv;
  X[(long)gw * 64 + lane] = a;       X[(long)gw * 64 + lane + 32] = b;
  qin[(long)gw * 64 + lane] = a;     qin[(long)gw * 64 + lane + 32] = b;
}

// codebook row squared-norms
__global__ void k_cnorm(const float* __restrict__ cb, float* __restrict__ cn) {
  int c = blockIdx.x * blockDim.x + threadIdx.x;
  if (c >= CBQ) return;
  float s = 0.f;
  #pragma unroll 8
  for (int d = 0; d < 64; d++) { float v = cb[(long)c * 64 + d]; s += v * v; }
  cn[c] = s;
}

// ---------------------------------------------------------------------------
// VQ argmin: score = |c|^2 - 2 x.c  (|x|^2 constant per row); codebook staged
// through LDS in 64-code tiles (16KB), broadcast reads across the block.
// ---------------------------------------------------------------------------
__global__ __launch_bounds__(128)
void k_vq(const float* __restrict__ X, const float* __restrict__ cb,
          const float* __restrict__ cn, int* __restrict__ idx) {
  __shared__ float sc[64 * 64];
  __shared__ float sn[64];
  int m = blockIdx.x * 128 + threadIdx.x;
  float xv[64];
  #pragma unroll
  for (int d = 0; d < 64; d++) xv[d] = X[(long)m * 64 + d];
  float best = 3.4e38f; int bi = 0;
  for (int c0 = 0; c0 < CBQ; c0 += 64) {
    __syncthreads();
    for (int i = threadIdx.x; i < 64 * 64; i += 128) sc[i] = cb[(long)c0 * 64 + i];
    if (threadIdx.x < 64) sn[threadIdx.x] = cn[c0 + threadIdx.x];
    __syncthreads();
    for (int c = 0; c < 64; c++) {
      float dot = 0.f;
      #pragma unroll
      for (int d = 0; d < 64; d++) dot += xv[d] * sc[c * 64 + d];
      float score = sn[c] - 2.f * dot;
      if (score < best) { best = score; bi = c0 + c; }
    }
  }
  idx[m] = bi;
}

// gather codebook rows -> X (decoder input) + quant_out region + indices
__global__ void k_gather(const int* __restrict__ idx, const float* __restrict__ cb,
                         float* __restrict__ X, float* __restrict__ qout,
                         float* __restrict__ indf) {
  int g = blockIdx.x * blockDim.x + threadIdx.x;    // exactly M*64 threads
  int m = g >> 6, d = g & 63;
  int c = idx[m];
  float v = cb[(long)c * 64 + d];
  X[g] = v; qout[g] = v;
  if (d == 0) indf[m] = (float)c;
}

// ---------------------------------------------------------------------------
// Host driver
// ---------------------------------------------------------------------------
extern "C" void kernel_launch(void* const* d_in, const int* in_sizes, int n_in,
                              void* d_out, int out_size, void* d_ws, size_t ws_size,
                              hipStream_t stream) {
  (void)in_sizes; (void)n_in; (void)out_size; (void)ws_size;

  const float* x       = (const float*)d_in[0];
  const float* enc_w   = (const float*)d_in[1];
  const float* enc_b   = (const float*)d_in[2];
  const float* e_ln1_s = (const float*)d_in[3];
  const float* e_ln1_b = (const float*)d_in[4];
  const float* e_qkv   = (const float*)d_in[5];
  const float* e_out   = (const float*)d_in[6];
  const float* e_ln2_s = (const float*)d_in[7];
  const float* e_ln2_b = (const float*)d_in[8];
  const float* e_f1w   = (const float*)d_in[9];
  const float* e_f1b   = (const float*)d_in[10];
  const float* e_f2w   = (const float*)d_in[11];
  const float* e_f2b   = (const float*)d_in[12];
  const float* d_ln1_s = (const float*)d_in[13];
  const float* d_ln1_b = (const float*)d_in[14];
  const float* d_qkv   = (const float*)d_in[15];
  const float* d_outw  = (const float*)d_in[16];
  const float* d_ln2_s = (const float*)d_in[17];
  const float* d_ln2_b = (const float*)d_in[18];
  const float* d_f1w   = (const float*)d_in[19];
  const float* d_f1b   = (const float*)d_in[20];
  const float* d_f2w   = (const float*)d_in[21];
  const float* d_f2b   = (const float*)d_in[22];
  const float* codebook= (const float*)d_in[23];
  const float* dec_w   = (const float*)d_in[24];
  const float* dec_b   = (const float*)d_in[25];

  float* out   = (float*)d_out;
  float* recon = out;                               // [B,NF,S] = 5242880
  float* ind_f = out + 5242880;                     // [B,S]    = 16384
  float* qout  = out + 5242880 + 16384;             // [B,S,D]  = 1048576
  float* qin   = qout + 1048576;                    // [B,S,D]  = 1048576

  // ---- workspace carve-up (256B aligned regions) ----
  char*  wsb = (char*)d_ws;
  size_t off = 0;
  auto alloc = [&](size_t bytes) -> void* {
    void* p = wsb + off;
    off += (bytes + 255) & ~(size_t)255;
    return p;
  };
  float*  X    = (float*)alloc((size_t)MQ * 64 * 4);
  float*  QKV  = (float*)alloc((size_t)MQ * 192 * 4);
  float*  KB   = (float*)alloc((size_t)MQ * 64 * 4);
  float*  CTX  = (float*)alloc((size_t)BQ * NHQ * 64 * 4);
  float*  CN   = (float*)alloc((size_t)CBQ * 4);
  int*    IDX  = (int*)  alloc((size_t)MQ * 4);
  __bf16* Ab   = (__bf16*)alloc((size_t)MQ * NFQ * 2);   // bf16 A-side staging
  __bf16* FB   = (__bf16*)alloc((size_t)MQ * HIDQ * 2);  // bf16 FFN hidden
  __bf16* enc_wt = (__bf16*)alloc((size_t)320 * 64 * 2);
  __bf16* eqkv_t = (__bf16*)alloc((size_t)LENCQ * 192 * 64 * 2);
  __bf16* eout_t = (__bf16*)alloc((size_t)LENCQ * 64 * 64 * 2);
  __bf16* ef1_t  = (__bf16*)alloc((size_t)LENCQ * 256 * 64 * 2);
  __bf16* ef2_t  = (__bf16*)alloc((size_t)LENCQ * 64 * 256 * 2);
  __bf16* dqkv_t = (__bf16*)alloc((size_t)LDECQ * 192 * 64 * 2);
  __bf16* dout_t = (__bf16*)alloc((size_t)LDECQ * 64 * 64 * 2);
  __bf16* df1_t  = (__bf16*)alloc((size_t)LDECQ * 256 * 64 * 2);
  __bf16* df2_t  = (__bf16*)alloc((size_t)LDECQ * 64 * 256 * 2);
  __bf16* dec_wt = (__bf16*)alloc((size_t)320 * 64 * 2);

  // ---- one-time weight conversion to transposed bf16 ----
  auto convw = [&](const float* w, __bf16* wt, int L, int K, int N) {
    long total = (long)L * K * N;
    int  blocks = (int)((total + 255) / 256);
    k_convw<<<blocks, 256, 0, stream>>>(w, wt, K, N, total);
  };
  convw(enc_w, enc_wt, 1, 320, 64);
  convw(e_qkv, eqkv_t, LENCQ, 64, 192);
  convw(e_out, eout_t, LENCQ, 64, 64);
  convw(e_f1w, ef1_t,  LENCQ, 64, 256);
  convw(e_f2w, ef2_t,  LENCQ, 256, 64);
  convw(d_qkv, dqkv_t, LDECQ, 64, 192);
  convw(d_outw, dout_t, LDECQ, 64, 64);
  convw(d_f1w, df1_t,  LDECQ, 64, 256);
  convw(d_f2w, df2_t,  LDECQ, 256, 64);
  convw(dec_w, dec_wt, 1, 64, 320);
  k_cnorm<<<CBQ / 256, 256, 0, stream>>>(codebook, CN);

  // ---- encoder input projection ----
  k_transpose_x<<<dim3(NFQ / 32, SQ / 32, BQ), dim3(32, 8), 0, stream>>>(x, Ab);
  k_gemm<EPI_BIAS, 320, 64><<<dim3(4, 256), 128, 0, stream>>>(Ab, enc_wt, enc_b, nullptr, X);
  k_addpe<<<MQ * 64 / 256, 256, 0, stream>>>(X);

  // ---- one transformer layer (pre-norm, linear attention, gelu FFN) ----
  auto layer = [&](const float* ln1s, const float* ln1b, const __bf16* qkvt,
                   const __bf16* outt, const float* ln2s, const float* ln2b,
                   const __bf16* f1t, const float* f1b,
                   const __bf16* f2t, const float* f2b) {
    k_layernorm<<<MQ / 8, 256, 0, stream>>>(X, ln1s, ln1b, Ab);
    k_gemm<EPI_NONE, 64, 192><<<dim3(12, 256), 128, 0, stream>>>(Ab, qkvt, nullptr, nullptr, QKV);
    k_ksoftmax<<<BQ * 64 / 8, 256, 0, stream>>>(QKV, KB);
    k_ctx<<<BQ * NHQ / 4, 256, 0, stream>>>(KB, QKV, CTX);
    k_qctx<<<MQ * 8 / 256, 256, 0, stream>>>(QKV, CTX, Ab);
    k_gemm<EPI_RESID, 64, 64><<<dim3(4, 256), 128, 0, stream>>>(Ab, outt, nullptr, X, X);
    k_layernorm<<<MQ / 8, 256, 0, stream>>>(X, ln2s, ln2b, Ab);
    k_gemm<EPI_BIAS_GELU_BF16, 64, 256><<<dim3(16, 256), 128, 0, stream>>>(Ab, f1t, f1b, nullptr, FB);
    k_gemm<EPI_BIAS_RESID, 256, 64><<<dim3(4, 256), 128, 0, stream>>>(FB, f2t, f2b, X, X);
  };

  for (int l = 0; l < LENCQ; l++)
    layer(e_ln1_s + l * 64, e_ln1_b + l * 64,
          eqkv_t + (size_t)l * 192 * 64, eout_t + (size_t)l * 64 * 64,
          e_ln2_s + l * 64, e_ln2_b + l * 64,
          ef1_t + (size_t)l * 256 * 64, e_f1b + l * 256,
          ef2_t + (size_t)l * 64 * 256, e_f2b + l * 64);

  // ---- quantization ----
  k_l2norm<<<MQ / 8, 256, 0, stream>>>(X, qin);
  k_vq<<<MQ / 128, 128, 0, stream>>>(X, codebook, CN, IDX);
  k_gather<<<MQ * 64 / 256, 256, 0, stream>>>(IDX, codebook, X, qout, ind_f);
  k_addpe<<<MQ * 64 / 256, 256, 0, stream>>>(X);

  // ---- decoder ----
  for (int l = 0; l < LDECQ; l++)
    layer(d_ln1_s + l * 64, d_ln1_b + l * 64,
          dqkv_t + (size_t)l * 192 * 64, dout_t + (size_t)l * 64 * 64,
          d_ln2_s + l * 64, d_ln2_b + l * 64,
          df1_t + (size_t)l * 256 * 64, d_f1b + l * 256,
          df2_t + (size_t)l * 64 * 256, d_f2b + l * 64);

  // ---- output projection + transposed store into recon [B,NF,S] ----
  k_convact<<<MQ * 64 / 256, 256, 0, stream>>>(X, Ab);
  k_gemm<EPI_BIAS_TRANS, 64, 320><<<dim3(20, 256), 128, 0, stream>>>(Ab, dec_wt, dec_b, nullptr, recon);
}
